// TransformerEncoderLayer_62423054680216
// MI455X (gfx1250) — matrix operand order repaired and checked by reading, not compiled, not run on hardware
//
#include <hip/hip_runtime.h>
#include <hip/hip_bf16.h>
#include <math.h>
#include <stdint.h>

// ---------------- problem constants ----------------
#define S_LEN 4096
#define D_DIM 768
#define H_CNT 12
#define HD_DIM 64
#define F_DIM 3072

typedef __attribute__((ext_vector_type(16))) __bf16 v16bf;
typedef __attribute__((ext_vector_type(8)))  float  v8f;

union BF16Frag { uint4 u[2]; v16bf v; };

// A-fragment (16x32, 16-bit): lane L holds row M=L%16; elems i -> k = 2*ks + i,
// where ks = (L>>4)*8 (caller passes ks; 16 contiguous elements along K).
__device__ __forceinline__ v16bf load_frag_a(const __bf16* rowp, int ks) {
  BF16Frag f;
  f.u[0] = *(const uint4*)(rowp + 2 * ks);
  f.u[1] = *(const uint4*)(rowp + 2 * ks + 8);
  return f.v;
}
// B-fragment (32x16, 16-bit): lane L holds col N=L%16; elems i -> k = (L>>4)*16 + i
// (caller passes ks2 = (L>>4)*16; 16 contiguous elements along K).
__device__ __forceinline__ v16bf load_frag_b(const __bf16* rowp, int ks2) {
  BF16Frag f;
  f.u[0] = *(const uint4*)(rowp + ks2);
  f.u[1] = *(const uint4*)(rowp + ks2 + 8);
  return f.v;
}

__device__ __forceinline__ v8f wmma_bf16(v16bf a, v16bf b, v8f c) {
  // D = A(16x32 bf16) * B(32x16 bf16) + C(16x16 f32)
  return __builtin_amdgcn_wmma_f32_16x16x32_bf16(false, a, false, b, (short)0, c, false, false);
}

// CDNA5 async DMA: global -> LDS, 16 bytes, tracked by ASYNCcnt (no VGPR round trip).
// LDS byte address = low 32 bits of the generic shared-memory pointer (ISA 10.2 aperture map).
__device__ __forceinline__ void async_copy_b128(void* lds_dst, const void* gsrc) {
  unsigned lds_addr = (unsigned)(uintptr_t)lds_dst;
  unsigned long long gaddr = (unsigned long long)(uintptr_t)gsrc;
  asm volatile("global_load_async_to_lds_b128 %0, %1, off"
               :: "v"(lds_addr), "v"(gaddr) : "memory");
}
__device__ __forceinline__ void wait_asynccnt0() {
  asm volatile("s_wait_asynccnt 0" ::: "memory");
}

// ---------------- fp32 -> bf16 conversions ----------------
__global__ void cvt_bf16_kernel(const float* __restrict__ in, __bf16* __restrict__ out, int n) {
  int i = blockIdx.x * 256 + threadIdx.x;
  if (i < n) out[i] = (__bf16)in[i];
}

// Wq/Wk/Wv are [H, D, HD]; repack to row-major [D, H*HD] so projections are plain GEMMs.
__global__ void cvt_qkvw_kernel(const float* __restrict__ in, __bf16* __restrict__ out) {
  int i = blockIdx.x * 256 + threadIdx.x; // over D*768
  if (i >= D_DIM * D_DIM) return;
  int d = i / D_DIM, c = i % D_DIM;
  int h = c >> 6, e = c & 63;
  out[i] = (__bf16)in[(size_t)h * D_DIM * HD_DIM + (size_t)d * HD_DIM + e];
}

// ---------------- LayerNorm (one wave32 per row of 768) ----------------
__global__ __launch_bounds__(256) void ln_bf16_kernel(const float* __restrict__ x,
                                                      const float* __restrict__ w,
                                                      const float* __restrict__ b,
                                                      __bf16* __restrict__ y) {
  const int lane = threadIdx.x;                       // 0..31
  const int row  = blockIdx.x * blockDim.y + threadIdx.y;
  const float* xr = x + (size_t)row * D_DIM;
  float xs[24];
  float sum = 0.f, sq = 0.f;
#pragma unroll
  for (int j = 0; j < 24; ++j) {
    float v = xr[lane + j * 32];
    xs[j] = v; sum += v; sq += v * v;
  }
#pragma unroll
  for (int m = 16; m >= 1; m >>= 1) {
    sum += __shfl_xor(sum, m, 32);
    sq  += __shfl_xor(sq,  m, 32);
  }
  float mu  = sum * (1.0f / D_DIM);
  float var = sq  * (1.0f / D_DIM) - mu * mu;
  float rs  = rsqrtf(var + 1e-5f);
  __bf16* yr = y + (size_t)row * D_DIM;
#pragma unroll
  for (int j = 0; j < 24; ++j) {
    int c = lane + j * 32;
    yr[c] = (__bf16)((xs[j] - mu) * rs * w[c] + b[c]);
  }
}

// ---------------- generic bf16 GEMM: C = act(A[M,K] @ B[K,N] + bias (+res)) ----------------
// Block: 256 threads (8 waves). Block tile 64x128, wave tile 32x32 (4 WMMA accumulators).
// A tile staged via async global->LDS DMA; B tile transposed in-flight via VGPRs.
__global__ __launch_bounds__(256) void gemm_bf16_kernel(
    const __bf16* __restrict__ A, const __bf16* __restrict__ B,
    const float* __restrict__ bias, const float* __restrict__ residual,
    float* __restrict__ outF, __bf16* __restrict__ outB,
    int M, int N, int K, int gelu) {
  __shared__ __bf16 As[64][32];    // [m][k]
  __shared__ __bf16 Bs[128][32];   // [n][k]  (transposed for contiguous B-frag reads)
  const int t = threadIdx.x;
  const int L = t & 31, w = t >> 5;
  const int bm = blockIdx.y * 64, bn = blockIdx.x * 128;
  const int wm = (w & 1) * 32,  wn = (w >> 1) * 32;
  const int l15 = L & 15, ks = (L >> 4) * 8, ks2 = (L >> 4) * 16;

  v8f acc00 = {}, acc01 = {}, acc10 = {}, acc11 = {};
  const int am = t >> 2, ak = (t & 3) * 8;   // A staging: 8 bf16 per thread
  for (int k0 = 0; k0 < K; k0 += 32) {
    // A tile: async DMA straight into LDS (ASYNCcnt), no VGPR round trip.
    async_copy_b128(&As[am][ak], &A[(size_t)(bm + am) * K + k0 + ak]);
#pragma unroll
    for (int j = 0; j < 2; ++j) {            // B staging w/ transpose: 2x8 bf16 per thread
      int c = t + 256 * j;
      int kk = c >> 4, nn = (c & 15) * 8;
      uint4 val = *(const uint4*)&B[(size_t)(k0 + kk) * N + bn + nn];
      const __bf16* pv = (const __bf16*)&val;
#pragma unroll
      for (int e = 0; e < 8; ++e) Bs[nn + e][kk] = pv[e];
    }
    // prefetch next K-step tiles into L2 while WMMAs run on this one
    if (k0 + 32 < K) {
      __builtin_prefetch(&A[(size_t)(bm + am) * K + k0 + 32 + ak], 0, 0);
      __builtin_prefetch(&B[(size_t)(k0 + 32 + (t >> 4)) * N + bn + (t & 15) * 8], 0, 0);
    }
    wait_asynccnt0();
    __syncthreads();
    v16bf a0 = load_frag_a(&As[wm + l15][0], ks);
    v16bf a1 = load_frag_a(&As[wm + 16 + l15][0], ks);
    v16bf b0 = load_frag_b(&Bs[wn + l15][0], ks2);
    v16bf b1 = load_frag_b(&Bs[wn + 16 + l15][0], ks2);
    acc00 = wmma_bf16(a0, b0, acc00);
    acc01 = wmma_bf16(a0, b1, acc01);
    acc10 = wmma_bf16(a1, b0, acc10);
    acc11 = wmma_bf16(a1, b1, acc11);
    __syncthreads();
  }
  v8f* accp[2][2] = {{&acc00, &acc01}, {&acc10, &acc11}};
#pragma unroll
  for (int mi = 0; mi < 2; ++mi)
#pragma unroll
    for (int ni = 0; ni < 2; ++ni) {
      v8f a = *accp[mi][ni];
      int col = bn + wn + ni * 16 + l15;
      float bs = bias[col];
#pragma unroll
      for (int r = 0; r < 8; ++r) {
        int row = bm + wm + mi * 16 + r + 8 * (L >> 4);
        size_t idx = (size_t)row * N + col;
        float v = a[r] + bs;
        if (residual) v += residual[idx];
        if (gelu) v = 0.5f * v * (1.0f + erff(v * 0.70710678118654752f));
        if (outF) outF[idx] = v;
        if (outB) outB[idx] = (__bf16)v;
      }
    }
}

// ---------------- fused flash attention ----------------
// Block: 256 threads = 8 waves; head = blockIdx.y, queries = blockIdx.x*128 .. +127
// (wave w owns 16 query rows). K tile async-DMA'd to LDS, V transposed via VGPRs;
// online softmax with per-row stats replicated across 16-lane halves.
__global__ __launch_bounds__(256) void attn_kernel(
    const __bf16* __restrict__ Q, const __bf16* __restrict__ Kb,
    const __bf16* __restrict__ V, __bf16* __restrict__ O) {
  __shared__ __bf16 Ks[32][64];       // [key][e]
  __shared__ __bf16 Vt[64][32];       // [e][key] (transposed)
  __shared__ __bf16 Ps[8][16][32];    // per-wave P tile for layout swap
  const int t = threadIdx.x, L = t & 31, w = t >> 5;
  const int h = blockIdx.y;
  const int qbase = blockIdx.x * 128 + w * 16;
  const int l15 = L & 15, ks8 = (L >> 4) * 8, ks16 = (L >> 4) * 16;

  const size_t qoff = (size_t)(qbase + l15) * D_DIM + h * HD_DIM;
  v16bf aq0 = load_frag_a(Q + qoff, ks8);        // e = 0..31
  v16bf aq1 = load_frag_a(Q + qoff + 32, ks8);   // e = 32..63

  float m_i[8], l_i[8];
#pragma unroll
  for (int r = 0; r < 8; ++r) { m_i[r] = -1e30f; l_i[r] = 0.f; }
  v8f o0 = {}, o1 = {}, o2 = {}, o3 = {};

  const int lkey = t >> 3, lec = (t & 7) * 8;    // staging indices
  for (int kb = 0; kb < S_LEN; kb += 32) {
    {
      // K tile: async global->LDS DMA (straight copy)
      async_copy_b128(&Ks[lkey][lec],
                      &Kb[(size_t)(kb + lkey) * D_DIM + h * HD_DIM + lec]);
      // V tile: load + transpose into LDS
      uint4 vv = *(const uint4*)&V[(size_t)(kb + lkey) * D_DIM + h * HD_DIM + lec];
      const __bf16* pv = (const __bf16*)&vv;
#pragma unroll
      for (int e = 0; e < 8; ++e) Vt[lec + e][lkey] = pv[e];
    }
    wait_asynccnt0();
    __syncthreads();

    // scores: S = Q[16x64] @ K^T[64x32]  (two 16x16 C tiles)
    v8f s0 = {}, s1 = {};
    {
      v16bf bk;
      bk = load_frag_b(&Ks[l15][0],      ks16);      s0 = wmma_bf16(aq0, bk, s0);
      bk = load_frag_b(&Ks[l15][0],      32 + ks16); s0 = wmma_bf16(aq1, bk, s0);
      bk = load_frag_b(&Ks[16 + l15][0], ks16);      s1 = wmma_bf16(aq0, bk, s1);
      bk = load_frag_b(&Ks[16 + l15][0], 32 + ks16); s1 = wmma_bf16(aq1, bk, s1);
    }

    // online softmax per row (row r+8*(L>>4) lives in 16 lanes of a half-wave)
#pragma unroll
    for (int r = 0; r < 8; ++r) {
      float v0 = s0[r] * 0.125f, v1 = s1[r] * 0.125f;  // 1/sqrt(64)
      float mx = fmaxf(v0, v1);
#pragma unroll
      for (int msk = 8; msk >= 1; msk >>= 1) mx = fmaxf(mx, __shfl_xor(mx, msk, 16));
      float mnew  = fmaxf(m_i[r], mx);
      float alpha = __expf(m_i[r] - mnew);
      float p0 = __expf(v0 - mnew), p1 = __expf(v1 - mnew);
      float rs = p0 + p1;
#pragma unroll
      for (int msk = 8; msk >= 1; msk >>= 1) rs += __shfl_xor(rs, msk, 16);
      l_i[r] = l_i[r] * alpha + rs;
      m_i[r] = mnew;
      o0[r] *= alpha; o1[r] *= alpha; o2[r] *= alpha; o3[r] *= alpha;
      int prow = r + 8 * (L >> 4);
      Ps[w][prow][l15]      = (__bf16)p0;
      Ps[w][prow][l15 + 16] = (__bf16)p1;
    }
    // wave-local DS RAW: drain DScnt before re-reading our own P tile
    asm volatile("s_wait_dscnt 0" ::: "memory");

    // O += P[16x32] @ V[32x64]  (four 16x16 C tiles along e)
    v16bf ap = load_frag_a(&Ps[w][l15][0], ks8);
    {
      v16bf bv;
      bv = load_frag_b(&Vt[l15][0],      ks16); o0 = wmma_bf16(ap, bv, o0);
      bv = load_frag_b(&Vt[16 + l15][0], ks16); o1 = wmma_bf16(ap, bv, o1);
      bv = load_frag_b(&Vt[32 + l15][0], ks16); o2 = wmma_bf16(ap, bv, o2);
      bv = load_frag_b(&Vt[48 + l15][0], ks16); o3 = wmma_bf16(ap, bv, o3);
    }
    __syncthreads();
  }

  v8f* op[4] = {&o0, &o1, &o2, &o3};
#pragma unroll
  for (int r = 0; r < 8; ++r) {
    float inv = 1.0f / l_i[r];
    int rowg = qbase + r + 8 * (L >> 4);
    size_t base = (size_t)rowg * D_DIM + h * HD_DIM + l15;
#pragma unroll
    for (int nt = 0; nt < 4; ++nt)
      O[base + nt * 16] = (__bf16)((*op[nt])[r] * inv);
  }
}

// ---------------- host-side orchestration ----------------
extern "C" void kernel_launch(void* const* d_in, const int* in_sizes, int n_in,
                              void* d_out, int out_size, void* d_ws, size_t ws_size,
                              hipStream_t stream) {
  (void)in_sizes; (void)n_in; (void)out_size; (void)ws_size;
  const float* x    = (const float*)d_in[0];
  const float* Wq   = (const float*)d_in[1];
  const float* bq   = (const float*)d_in[2];
  const float* Wk   = (const float*)d_in[3];
  const float* bk   = (const float*)d_in[4];
  const float* Wv   = (const float*)d_in[5];
  const float* bv   = (const float*)d_in[6];
  const float* Wo   = (const float*)d_in[7];
  const float* bo   = (const float*)d_in[8];
  const float* ln1w = (const float*)d_in[9];
  const float* ln1b = (const float*)d_in[10];
  const float* ln2w = (const float*)d_in[11];
  const float* ln2b = (const float*)d_in[12];
  const float* W1   = (const float*)d_in[13];
  const float* b1   = (const float*)d_in[14];
  const float* W2   = (const float*)d_in[15];
  const float* b2   = (const float*)d_in[16];

  char* ws = (char*)d_ws;
  size_t off = 0;
  auto wsalloc = [&](size_t bytes) -> char* {
    char* p = ws + off; off += (bytes + 255) & ~(size_t)255; return p;
  };
  const size_t SD2 = (size_t)S_LEN * D_DIM * 2;
  __bf16* Ybf  = (__bf16*)wsalloc(SD2);                       // LN1(x)
  __bf16* Qbf  = (__bf16*)wsalloc(SD2);
  __bf16* Kbf  = (__bf16*)wsalloc(SD2);
  __bf16* Vbf  = (__bf16*)wsalloc(SD2);
  __bf16* Obf  = (__bf16*)wsalloc(SD2);                       // concat attention out
  float*  X1   = (float*)wsalloc((size_t)S_LEN * D_DIM * 4);  // x + attn
  __bf16* Y2bf = (__bf16*)wsalloc(SD2);                       // LN2(X1)
  __bf16* Hbf  = (__bf16*)wsalloc((size_t)S_LEN * F_DIM * 2); // gelu(ffn1)
  __bf16* Wqb  = (__bf16*)wsalloc((size_t)D_DIM * D_DIM * 2);
  __bf16* Wkb  = (__bf16*)wsalloc((size_t)D_DIM * D_DIM * 2);
  __bf16* Wvb  = (__bf16*)wsalloc((size_t)D_DIM * D_DIM * 2);
  __bf16* Wob  = (__bf16*)wsalloc((size_t)D_DIM * D_DIM * 2);
  __bf16* W1b  = (__bf16*)wsalloc((size_t)D_DIM * F_DIM * 2);
  __bf16* W2b  = (__bf16*)wsalloc((size_t)F_DIM * D_DIM * 2);

  const int nDD = D_DIM * D_DIM, nDF = D_DIM * F_DIM;
  // weight conversions
  cvt_qkvw_kernel<<<(nDD + 255) / 256, 256, 0, stream>>>(Wq, Wqb);
  cvt_qkvw_kernel<<<(nDD + 255) / 256, 256, 0, stream>>>(Wk, Wkb);
  cvt_qkvw_kernel<<<(nDD + 255) / 256, 256, 0, stream>>>(Wv, Wvb);
  cvt_bf16_kernel<<<(nDD + 255) / 256, 256, 0, stream>>>(Wo, Wob, nDD);
  cvt_bf16_kernel<<<(nDF + 255) / 256, 256, 0, stream>>>(W1, W1b, nDF);
  cvt_bf16_kernel<<<(nDF + 255) / 256, 256, 0, stream>>>(W2, W2b, nDF);

  // LN1
  ln_bf16_kernel<<<dim3(S_LEN / 8), dim3(32, 8), 0, stream>>>(x, ln1w, ln1b, Ybf);

  // QKV projections (M=4096, N=768, K=768)
  dim3 gP(D_DIM / 128, S_LEN / 64);
  gemm_bf16_kernel<<<gP, 256, 0, stream>>>(Ybf, Wqb, bq, nullptr, nullptr, Qbf,
                                           S_LEN, D_DIM, D_DIM, 0);
  gemm_bf16_kernel<<<gP, 256, 0, stream>>>(Ybf, Wkb, bk, nullptr, nullptr, Kbf,
                                           S_LEN, D_DIM, D_DIM, 0);
  gemm_bf16_kernel<<<gP, 256, 0, stream>>>(Ybf, Wvb, bv, nullptr, nullptr, Vbf,
                                           S_LEN, D_DIM, D_DIM, 0);

  // fused attention
  attn_kernel<<<dim3(S_LEN / 128, H_CNT), 256, 0, stream>>>(Qbf, Kbf, Vbf, Obf);

  // output projection + residual -> X1 (f32)
  gemm_bf16_kernel<<<gP, 256, 0, stream>>>(Obf, Wob, bo, x, X1, nullptr,
                                           S_LEN, D_DIM, D_DIM, 0);

  // LN2
  ln_bf16_kernel<<<dim3(S_LEN / 8), dim3(32, 8), 0, stream>>>(X1, ln2w, ln2b, Y2bf);

  // FFN1 + exact GELU (M=4096, N=3072, K=768)
  dim3 gF1(F_DIM / 128, S_LEN / 64);
  gemm_bf16_kernel<<<gF1, 256, 0, stream>>>(Y2bf, W1b, b1, nullptr, nullptr, Hbf,
                                            S_LEN, F_DIM, D_DIM, 1);

  // FFN2 + residual -> d_out (f32)  (M=4096, N=768, K=3072)
  gemm_bf16_kernel<<<gP, 256, 0, stream>>>(Hbf, W2b, b2, X1, (float*)d_out, nullptr,
                                           S_LEN, D_DIM, F_DIM, 0);
}
